// model_case_1_78400333021574
// MI455X (gfx1250) — matrix operand order repaired and checked by reading, not compiled, not run on hardware
//
#include <hip/hip_runtime.h>
#include <hip/hip_bf16.h>

#define NN 2048
#define DD 128
#define NEGV -1000000000.0f
#define LALPHA 0.2f
#define PSTRIDE 40   // padded halfs per LDS panel row (80B: 16B-aligned, spreads banks)

typedef _Float16 hx8  __attribute__((ext_vector_type(8)));
typedef _Float16 hx16 __attribute__((ext_vector_type(16)));
typedef float    fx8  __attribute__((ext_vector_type(8)));

__device__ __forceinline__ hx16 cat16(hx8 lo, hx8 hi) {
  return __builtin_shufflevector(lo, hi, 0,1,2,3,4,5,6,7,8,9,10,11,12,13,14,15);
}
__device__ __forceinline__ fx8 wmma16(hx16 a, hx16 b, fx8 c) {
  return __builtin_amdgcn_wmma_f32_16x16x32_f16(false, a, false, b, (short)0, c, false, false);
}
__device__ __forceinline__ float sigf(float x) { return 1.0f / (1.0f + __expf(-x)); }

// CDNA5 async copy: per-lane 16B global -> LDS, tracked by ASYNCcnt.
__device__ __forceinline__ void async_b128(uint32_t lds_off, const void* gaddr) {
  asm volatile("global_load_async_to_lds_b128 %0, %1, off"
               :: "v"(lds_off), "v"(gaddr) : "memory");
}
__device__ __forceinline__ void wait_async_le4() {
  asm volatile("s_wait_asynccnt 0x4" ::: "memory");
}
__device__ __forceinline__ void wait_async_0() {
  asm volatile("s_wait_asynccnt 0x0" ::: "memory");
}

// ---------------------------------------------------------------- prep ------
__global__ void k_prep_wg(const float* __restrict__ Wg, _Float16* __restrict__ WgT) {
  int idx = blockIdx.x * blockDim.x + threadIdx.x;
  if (idx >= 4 * 256 * 128) return;
  int g = idx / (256 * 128);
  int r = idx - g * 256 * 128;
  int f = r / 128, o = r % 128;
  WgT[g * 32768 + o * 256 + f] = (_Float16)Wg[g * 32768 + f * 128 + o];
}
__global__ void k_prep_w(const float* __restrict__ W, _Float16* __restrict__ WT) {
  int idx = blockIdx.x * blockDim.x + threadIdx.x;
  if (idx >= 128 * 128) return;
  int o = idx / 128, c = idx % 128;
  WT[o * 128 + c] = (_Float16)W[c * 128 + o];
}
__global__ void k_init_state(const float* __restrict__ actors, const float* __restrict__ h0,
                             float* __restrict__ cell, _Float16* __restrict__ hidden_h) {
  int idx = blockIdx.x * blockDim.x + threadIdx.x;
  if (idx >= NN * DD) return;
  int n = idx / DD, d = idx % DD;
  cell[idx] = actors[n * 20 * DD + d];
  hidden_h[idx] = (_Float16)h0[d];
}
__global__ void k_build_gin(const float* __restrict__ actors, const _Float16* __restrict__ hidden_h,
                            _Float16* __restrict__ gin, int x_idx) {
  int idx = blockIdx.x * blockDim.x + threadIdx.x;
  if (idx >= NN * 256) return;
  int n = idx / 256, f = idx % 256;
  gin[idx] = (f < DD) ? (_Float16)actors[n * 20 * DD + x_idx * DD + f]
                      : hidden_h[n * DD + (f - DD)];
}

// -------------------------------------------------- Wh = gin @ Wg (WMMA) ----
// block = 4 waves (4 row tiles, same gate). B k-panels (32x128 f16) staged in
// LDS via async-to-LDS, double-buffered; each wave computes 16x128.
__global__ void __launch_bounds__(128) k_wh_gemm(const _Float16* __restrict__ gin,
                                                 const _Float16* __restrict__ WgT,
                                                 float* __restrict__ Wh,
                                                 _Float16* __restrict__ WhT) {
  __shared__ _Float16 panel[2][DD * PSTRIDE];           // 2 x 10KB
  int tid = threadIdx.x, l = tid & 31, w = tid >> 5;
  int rt = blockIdx.x * 4 + w, g = blockIdx.y;
  int lrow = l & 15, kh = l >> 4;
  int row = rt * 16 + lrow;
  const _Float16* wsrc = WgT + g * 32768 + tid * 256;   // row o = tid
  auto stage = [&](int kbase, int buf) {                // 64B of row o per thread
    const _Float16* src = wsrc + kbase;
    uint32_t dst = (uint32_t)(uintptr_t)&panel[buf][tid * PSTRIDE];
    async_b128(dst, src);           async_b128(dst + 16, src + 8);
    async_b128(dst + 32, src + 16); async_b128(dst + 48, src + 24);
  };
  stage(0, 0);
  fx8 acc[8] = {};
  for (int kk = 0; kk < 8; ++kk) {
    int kbase = kk * 32, buf = kk & 1;
    if (kk < 7) { stage(kbase + 32, buf ^ 1); wait_async_le4(); }
    else        { wait_async_0(); }
    __syncthreads();                                    // panel[buf] ready for all waves
    const _Float16* ga = gin + row * 256 + kbase;
    hx16 A = cat16(*(const hx8*)(ga + kh * 16), *(const hx8*)(ga + kh * 16 + 8));
#pragma unroll
    for (int ct = 0; ct < 8; ++ct) {
      const _Float16* pb = &panel[buf][(ct * 16 + lrow) * PSTRIDE + kh * 16];
      acc[ct] = wmma16(A, cat16(*(const hx8*)pb, *(const hx8*)(pb + 8)), acc[ct]);
    }
    __syncthreads();                                    // protect buf before re-stage
  }
  int mb = rt * 16 + kh * 8;
#pragma unroll
  for (int ct = 0; ct < 8; ++ct) {
    int o = ct * 16 + lrow;
    hx8 hv;
#pragma unroll
    for (int r = 0; r < 8; ++r) {
      float v = acc[ct][r];
      Wh[g * NN * DD + (mb + r) * DD + o] = v;
      hv[r] = (_Float16)v;
    }
    *(hx8*)(WhT + g * DD * NN + o * NN + mb) = hv;      // transposed f16 copy
  }
}

// -------------------------------------- s_src / s_dst = Wh . ag halves ------
__global__ void k_svec(const float* __restrict__ Wh, const float* __restrict__ ag,
                       float* __restrict__ s_src, float* __restrict__ s_dst) {
  int n = blockIdx.x, g = blockIdx.y, l = threadIdx.x;
  float a = 0.f, b = 0.f;
  for (int d = l; d < DD; d += 32) {
    float wv = Wh[g * NN * DD + n * DD + d];
    a += wv * ag[g * 256 + d];
    b += wv * ag[g * 256 + 128 + d];
  }
  for (int off = 16; off; off >>= 1) { a += __shfl_xor(a, off, 32); b += __shfl_xor(b, off, 32); }
  if (l == 0) { s_src[g * NN + n] = a; s_dst[g * NN + n] = b; }
}

// --------------------------- online softmax stats (max, 1/sum) per row ------
__global__ void k_stats(const int* __restrict__ adjs, const float* __restrict__ s_src,
                        const float* __restrict__ s_dst, float* __restrict__ rmax,
                        float* __restrict__ rinv, int e_idx) {
  int n = blockIdx.x, l = threadIdx.x;
  const int* adjrow = adjs + (size_t)n * 20 * NN + e_idx * NN;
  float m[4], s[4], ss[4];
#pragma unroll
  for (int g = 0; g < 4; ++g) { m[g] = -INFINITY; s[g] = 0.f; ss[g] = s_src[g * NN + n]; }
  for (int c = l; c < NN; c += 32) {
    __builtin_prefetch(adjrow + c + 256, 0, 1);
    bool mk = adjrow[c] > 0;
#pragma unroll
    for (int g = 0; g < 4; ++g) {
      float e;
      if (mk) { float x = ss[g] + s_dst[g * NN + c]; e = x > 0.f ? x : LALPHA * x; }
      else e = NEGV;
      if (e > m[g]) { s[g] = s[g] * __expf(m[g] - e) + 1.0f; m[g] = e; }
      else s[g] += __expf(e - m[g]);
    }
  }
  for (int off = 16; off; off >>= 1) {
#pragma unroll
    for (int g = 0; g < 4; ++g) {
      float mo = __shfl_xor(m[g], off, 32), so = __shfl_xor(s[g], off, 32);
      float mn = fmaxf(m[g], mo);
      s[g] = s[g] * __expf(m[g] - mn) + so * __expf(mo - mn);
      m[g] = mn;
    }
  }
  if (l == 0)
#pragma unroll
    for (int g = 0; g < 4; ++g) { rmax[g * NN + n] = m[g]; rinv[g * NN + n] = 1.0f / s[g]; }
}

// ------------------- gates[g] = elu( softmax(e) @ Wh[g] )  (flash, WMMA) ----
// block = 4 waves sharing the streamed WhT k-panel in LDS (async, dbl-buffer).
__global__ void __launch_bounds__(128) k_attn_av(const int* __restrict__ adjs,
                                                 const float* __restrict__ s_src,
                                                 const float* __restrict__ s_dst,
                                                 const float* __restrict__ rmax,
                                                 const float* __restrict__ rinv,
                                                 const _Float16* __restrict__ WhT,
                                                 float* __restrict__ gates, int e_idx) {
  __shared__ _Float16 panel[2][DD * PSTRIDE];           // 2 x 10KB
  int tid = threadIdx.x, l = tid & 31, w = tid >> 5;
  int rt = blockIdx.x * 4 + w, g = blockIdx.y;
  int lrow = l & 15, kh = l >> 4;
  int row = rt * 16 + lrow;
  float ssrc = s_src[g * NN + row];
  float mx = rmax[g * NN + row];
  float inv = rinv[g * NN + row];
  const int* adjrow = adjs + (size_t)row * 20 * NN + e_idx * NN;
  const float* sd = s_dst + g * NN;
  const _Float16* gbase = WhT + g * DD * NN + tid * NN; // panel row o = tid
  auto stage = [&](int m0, int buf) {
    const _Float16* src = gbase + m0;
    uint32_t dst = (uint32_t)(uintptr_t)&panel[buf][tid * PSTRIDE];
    async_b128(dst, src);           async_b128(dst + 16, src + 8);
    async_b128(dst + 32, src + 16); async_b128(dst + 48, src + 24);
  };
  stage(0, 0);
  fx8 acc[8] = {};
  for (int m0 = 0; m0 < NN; m0 += 32) {
    int buf = (m0 >> 5) & 1;
    if (m0 + 32 < NN) { stage(m0 + 32, buf ^ 1); wait_async_le4(); }
    else              { wait_async_0(); }
    __syncthreads();
    __builtin_prefetch(adjrow + m0 + 64, 0, 1);
    hx16 A;
#pragma unroll
    for (int j = 0; j < 8; ++j) {
      int c1 = m0 + kh * 16 + j;
      int c2 = c1 + 8;
      float e1 = NEGV, e2 = NEGV;
      if (adjrow[c1] > 0) { float x = ssrc + sd[c1]; e1 = x > 0.f ? x : LALPHA * x; }
      if (adjrow[c2] > 0) { float x = ssrc + sd[c2]; e2 = x > 0.f ? x : LALPHA * x; }
      A[j]     = (_Float16)(__expf(e1 - mx) * inv);
      A[8 + j] = (_Float16)(__expf(e2 - mx) * inv);
    }
#pragma unroll
    for (int ct = 0; ct < 8; ++ct) {
      const _Float16* pb = &panel[buf][(ct * 16 + lrow) * PSTRIDE + kh * 16];
      acc[ct] = wmma16(A, cat16(*(const hx8*)pb, *(const hx8*)(pb + 8)), acc[ct]);
    }
    __syncthreads();
  }
  int mb = rt * 16 + kh * 8;
#pragma unroll
  for (int ct = 0; ct < 8; ++ct) {
    int o = ct * 16 + lrow;
#pragma unroll
    for (int r = 0; r < 8; ++r) {
      float v = acc[ct][r];
      v = v > 0.f ? v : (__expf(v) - 1.0f);               // ELU
      gates[g * NN * DD + (mb + r) * DD + o] = v;
    }
  }
}

// ---------------------------------------------------------- LSTM update -----
__global__ void k_lstm(const float* __restrict__ gates, float* __restrict__ cell,
                       _Float16* __restrict__ hidden_h) {
  int idx = blockIdx.x * blockDim.x + threadIdx.x;
  if (idx >= NN * DD) return;
  const int ND = NN * DD;
  float f = sigf(gates[idx]);
  float i = sigf(gates[ND + idx]);
  float cc = tanhf(gates[2 * ND + idx]);
  float o = sigf(gates[3 * ND + idx]);
  float c = cell[idx] * f + i * cc;
  cell[idx] = c;
  hidden_h[idx] = (_Float16)(tanhf(c) * o);
}

// ------------------------- y[:,:,t] = sigmoid(hidden @ W + b)  (WMMA) -------
__global__ void k_outproj(const _Float16* __restrict__ hidden_h, const _Float16* __restrict__ WT,
                          const float* __restrict__ bias, float* __restrict__ y, int t) {
  int l = threadIdx.x, rt = blockIdx.x;
  int lrow = l & 15, kh = l >> 4;
  int row = rt * 16 + lrow;
  fx8 acc[8] = {};
#pragma unroll
  for (int kk = 0; kk < 4; ++kk) {
    int kbase = kk * 32;
    const _Float16* ga = hidden_h + row * DD + kbase;
    hx16 A = cat16(*(const hx8*)(ga + kh * 16), *(const hx8*)(ga + kh * 16 + 8));
#pragma unroll
    for (int ct = 0; ct < 8; ++ct) {
      int o = ct * 16 + lrow;
      const _Float16* gb = WT + o * DD + kbase + kh * 16;
      acc[ct] = wmma16(A, cat16(*(const hx8*)gb, *(const hx8*)(gb + 8)), acc[ct]);
    }
  }
  int mb = rt * 16 + kh * 8;
#pragma unroll
  for (int ct = 0; ct < 8; ++ct) {
    int o = ct * 16 + lrow;
    float bv = bias[o];
#pragma unroll
    for (int r = 0; r < 8; ++r)
      y[(mb + r) * (DD * 4) + o * 4 + t] = sigf(acc[ct][r] + bv);
  }
}

// --------------------------- temporal conv layer: (N,C,Tin)->(N,C,Tin-1) ----
__global__ void k_conv(const float* __restrict__ in, float* __restrict__ out,
                       const float* __restrict__ cw, const float* __restrict__ cb,
                       int T_in) {
  int n = blockIdx.x, tt = blockIdx.y, o = threadIdx.x;
  int T_out = T_in - 1;
  __shared__ float y0s[DD], y1s[DD];
  y0s[o] = in[n * DD * T_in + o * T_in + tt];
  y1s[o] = in[n * DD * T_in + o * T_in + tt + 1];
  __syncthreads();
  const float* wv = cw + o * 256;
  float s = cb[o];
#pragma unroll 4
  for (int c = 0; c < DD; ++c)
    s += y0s[c] * wv[c * 2] + y1s[c] * wv[c * 2 + 1];
  out[n * DD * T_out + o * T_out + tt] = s;
}

// ---------------------------------------------------------------------------
extern "C" void kernel_launch(void* const* d_in, const int* in_sizes, int n_in,
                              void* d_out, int out_size, void* d_ws, size_t ws_size,
                              hipStream_t stream) {
  const float* actors = (const float*)d_in[0];
  const int*   adjs   = (const int*)d_in[1];
  const float* Wg     = (const float*)d_in[2];
  const float* ag     = (const float*)d_in[3];
  const float* W      = (const float*)d_in[4];
  const float* bias   = (const float*)d_in[5];
  const float* h0     = (const float*)d_in[6];
  const float* convw  = (const float*)d_in[7];
  const float* convb  = (const float*)d_in[8];
  float* out = (float*)d_out;

  size_t off = 0;
  auto alloc = [&](size_t bytes) {
    void* p = (char*)d_ws + off;
    off += (bytes + 255) & ~(size_t)255;
    return p;
  };
  _Float16* gin      = (_Float16*)alloc((size_t)NN * 256 * 2);
  _Float16* WgT      = (_Float16*)alloc((size_t)4 * 128 * 256 * 2);
  _Float16* WT       = (_Float16*)alloc((size_t)128 * 128 * 2);
  float*    Wh       = (float*)alloc((size_t)4 * NN * DD * 4);
  _Float16* WhT      = (_Float16*)alloc((size_t)4 * DD * NN * 2);
  float*    s_src    = (float*)alloc((size_t)4 * NN * 4);
  float*    s_dst    = (float*)alloc((size_t)4 * NN * 4);
  float*    rmax     = (float*)alloc((size_t)4 * NN * 4);
  float*    rinv     = (float*)alloc((size_t)4 * NN * 4);
  float*    gates    = (float*)alloc((size_t)4 * NN * DD * 4);
  float*    cell     = (float*)alloc((size_t)NN * DD * 4);
  _Float16* hidden_h = (_Float16*)alloc((size_t)NN * DD * 2);
  float*    y0       = (float*)alloc((size_t)NN * DD * 4 * 4);
  float*    y1       = (float*)alloc((size_t)NN * DD * 3 * 4);
  float*    y2       = (float*)alloc((size_t)NN * DD * 2 * 4);

  const int X_IDX[4] = {4, 9, 14, 19};
  const int E_IDX[4] = {0, 4, 9, 14};

  k_prep_wg<<<(4 * 256 * 128 + 255) / 256, 256, 0, stream>>>(Wg, WgT);
  k_prep_w<<<(128 * 128 + 255) / 256, 256, 0, stream>>>(W, WT);
  k_init_state<<<(NN * DD + 255) / 256, 256, 0, stream>>>(actors, h0, cell, hidden_h);

  for (int t = 0; t < 4; ++t) {
    k_build_gin<<<(NN * 256 + 255) / 256, 256, 0, stream>>>(actors, hidden_h, gin, X_IDX[t]);
    k_wh_gemm<<<dim3(NN / 64, 4), 128, 0, stream>>>(gin, WgT, Wh, WhT);
    k_svec<<<dim3(NN, 4), 32, 0, stream>>>(Wh, ag, s_src, s_dst);
    k_stats<<<NN, 32, 0, stream>>>(adjs, s_src, s_dst, rmax, rinv, E_IDX[t]);
    k_attn_av<<<dim3(NN / 64, 4), 128, 0, stream>>>(adjs, s_src, s_dst, rmax, rinv, WhT, gates,
                                                    E_IDX[t]);
    k_lstm<<<(NN * DD + 255) / 256, 256, 0, stream>>>(gates, cell, hidden_h);
    k_outproj<<<NN / 16, 32, 0, stream>>>(hidden_h, WT, bias, y0, t);
  }

  k_conv<<<dim3(NN, 3), DD, 0, stream>>>(y0, y1, convw + 0 * 32768, convb + 0 * 128, 4);
  k_conv<<<dim3(NN, 2), DD, 0, stream>>>(y1, y2, convw + 1 * 32768, convb + 1 * 128, 3);
  k_conv<<<dim3(NN, 1), DD, 0, stream>>>(y2, out, convw + 2 * 32768, convb + 2 * 128, 2);
}
